// DigitCaps_40209483825197
// MI455X (gfx1250) — compile-verified, matrix operand order checked by reading, not run on hardware
//
#include <hip/hip_runtime.h>
#include <hip/hip_bf16.h>

// ---------------------------------------------------------------------------
// DigitCaps dynamic routing for MI455X (gfx1250, wave32, WMMA).
//
// Never materializes u_hat (377 MB). Each routing step is re-expressed as an
// f16 WMMA GEMM over L2-resident operands (x: 9.4MB f16, cW: 2.9MB f16):
//   s_j   : (512 x 160 x 9216) GEMM, B-matrix = c[i,o]*W[i,o,p,s]
//   agree : (9216 x 160 x 512) GEMM G = x^T v, epilogue contracts with W
// ---------------------------------------------------------------------------

#define IN_CAPS   1152
#define S_DIM     8
#define OUT_CAPS  10
#define P_DIM     16
#define BATCH     512
#define KTOT      (IN_CAPS * S_DIM)    // 9216
#define NTOT      (OUT_CAPS * P_DIM)   // 160
#define KSPLIT    8                    // split-K chunks for s-GEMM (deterministic)

typedef _Float16 v8h  __attribute__((ext_vector_type(8)));
typedef _Float16 v16h __attribute__((ext_vector_type(16)));
typedef float    v8f  __attribute__((ext_vector_type(8)));

// Load one 16-half WMMA fragment for this lane: two contiguous 16B chunks.
// Matches the ISA 16-bit A/B layout: VGPRs 0-3 hold K[khalf..khalf+7],
// VGPRs 4-7 hold K[khalf+16..khalf+23].
__device__ __forceinline__ v16h load_frag(const _Float16* __restrict__ p) {
    v8h lo = *(const v8h*)(p);
    v8h hi = *(const v8h*)(p + 16);
    v16h r;
#pragma unroll
    for (int i = 0; i < 8; ++i) { r[i] = lo[i]; r[i + 8] = hi[i]; }
    return r;
}

// ---------------------------------------------------------------------------
// Utility kernels
// ---------------------------------------------------------------------------
__global__ void k_zero(float* __restrict__ p, int n) {
    int i = blockIdx.x * blockDim.x + threadIdx.x;
    if (i < n) p[i] = 0.0f;
}

// x (B,I,S) f32 -> xh[b][k] f16 (row-major, K contiguous) and xt[k][b] f16.
__global__ void k_convert_x(const float* __restrict__ x,
                            _Float16* __restrict__ xh,
                            _Float16* __restrict__ xt) {
    size_t idx = (size_t)blockIdx.x * blockDim.x + threadIdx.x;
    if (idx >= (size_t)BATCH * KTOT) return;
    int k = (int)(idx % KTOT);
    int b = (int)(idx / KTOT);
    _Float16 v = (_Float16)x[idx];
    xh[idx] = v;
    xt[(size_t)k * BATCH + b] = v;
}

// Column-wise softmax over i (axis 0) of bij (I,O). One block per o.
__global__ void k_softmax(const float* __restrict__ bij, float* __restrict__ c) {
    int o = blockIdx.x;
    __shared__ float red[256];
    float mx = -1e30f;
    for (int i = threadIdx.x; i < IN_CAPS; i += 256)
        mx = fmaxf(mx, bij[i * OUT_CAPS + o]);
    red[threadIdx.x] = mx; __syncthreads();
    for (int s = 128; s > 0; s >>= 1) {
        if (threadIdx.x < s) red[threadIdx.x] = fmaxf(red[threadIdx.x], red[threadIdx.x + s]);
        __syncthreads();
    }
    mx = red[0]; __syncthreads();
    float sm = 0.0f;
    for (int i = threadIdx.x; i < IN_CAPS; i += 256)
        sm += expf(bij[i * OUT_CAPS + o] - mx);
    red[threadIdx.x] = sm; __syncthreads();
    for (int s = 128; s > 0; s >>= 1) {
        if (threadIdx.x < s) red[threadIdx.x] += red[threadIdx.x + s];
        __syncthreads();
    }
    float inv = 1.0f / red[0];
    for (int i = threadIdx.x; i < IN_CAPS; i += 256)
        c[i * OUT_CAPS + o] = expf(bij[i * OUT_CAPS + o] - mx) * inv;
}

// cwT[(o*16+p)][(i*8+s)] = f16( c[i,o] * W[i,o,p,s] )   (B-matrix, K contiguous)
__global__ void k_prep_cw(const float* __restrict__ c, const float* __restrict__ W,
                          _Float16* __restrict__ cwT) {
    size_t idx = (size_t)blockIdx.x * blockDim.x + threadIdx.x;
    if (idx >= (size_t)NTOT * KTOT) return;
    int k = (int)(idx % KTOT);
    int n = (int)(idx / KTOT);
    int i = k >> 3, s = k & 7;
    int o = n >> 4, p = n & 15;
    float w = W[(((size_t)i * OUT_CAPS + o) * P_DIM + p) * S_DIM + s];
    cwT[idx] = (_Float16)(c[i * OUT_CAPS + o] * w);
}

// ---------------------------------------------------------------------------
// s-GEMM: s[b][(o,p)] = sum_k xh[b][k] * cwT[(o,p)][k],  M=512 N=160 K=9216.
// grid = (32 M-tiles, KSPLIT), block = 320 (10 waves, wave w -> o = w).
// Writes deterministic per-chunk partials.
// ---------------------------------------------------------------------------
__global__ void k_s_gemm(const _Float16* __restrict__ xh,
                         const _Float16* __restrict__ cwT,
                         float* __restrict__ spart) {
    const int mtile  = blockIdx.x;
    const int kchunk = blockIdx.y;
    const int wave   = threadIdx.x >> 5;   // o
    const int lane   = threadIdx.x & 31;
    const int nm     = lane & 15;          // row (A) / col (B) within tile
    const int khalf  = (lane >> 4) * 8;

    const int kbase  = kchunk * (KTOT / KSPLIT);
    const int ksteps = KTOT / 32 / KSPLIT;  // 36

    const _Float16* arow = xh  + (size_t)(mtile * 16 + nm) * KTOT + kbase + khalf;
    const _Float16* brow = cwT + (size_t)(wave  * 16 + nm) * KTOT + kbase + khalf;

    v8f acc = {};
    for (int ks = 0; ks < ksteps; ++ks) {
        __builtin_prefetch(arow + 64, 0, 1);   // global_prefetch_b8
        __builtin_prefetch(brow + 64, 0, 1);
        v16h a = load_frag(arow);
        v16h b = load_frag(brow);
        acc = __builtin_amdgcn_wmma_f32_16x16x32_f16(
            false, a, false, b, (short)0, acc, false, false);
        arow += 32; brow += 32;
    }

    // C layout: VGPR r, lanes 0-15 -> M=r, N=lane; lanes 16-31 -> M=8+r.
    const int mrow = (lane >> 4) * 8;
    float* out = spart + (size_t)kchunk * BATCH * NTOT;
#pragma unroll
    for (int r = 0; r < 8; ++r)
        out[(size_t)(mtile * 16 + mrow + r) * NTOT + wave * 16 + nm] = acc[r];
}

__global__ void k_reduce_s(const float* __restrict__ spart, float* __restrict__ s) {
    int idx = blockIdx.x * blockDim.x + threadIdx.x;
    if (idx >= BATCH * NTOT) return;
    float acc = 0.0f;
#pragma unroll
    for (int kc = 0; kc < KSPLIT; ++kc)
        acc += spart[(size_t)kc * BATCH * NTOT + idx];
    s[idx] = acc;
}

// squash per (b,o); writes vhT[(o,p)][b] f16 (agree-GEMM B operand) and,
// on the last iteration, f32 output (B,10,16).
__global__ void k_squash(const float* __restrict__ s, _Float16* __restrict__ vhT,
                         float* __restrict__ vout, int writeOut) {
    int idx = blockIdx.x * blockDim.x + threadIdx.x;
    if (idx >= BATCH * OUT_CAPS) return;
    int b = idx / OUT_CAPS, o = idx % OUT_CAPS;
    const float* sp = s + (size_t)b * NTOT + o * 16;
    float vals[16], sq = 0.0f;
#pragma unroll
    for (int p = 0; p < 16; ++p) { vals[p] = sp[p]; sq += vals[p] * vals[p]; }
    float scale = (sq / (1.0f + sq)) * rsqrtf(sq + 1e-8f);
#pragma unroll
    for (int p = 0; p < 16; ++p) {
        float v = vals[p] * scale;
        vhT[(size_t)(o * 16 + p) * BATCH + b] = (_Float16)v;
        if (writeOut) vout[(size_t)b * NTOT + o * 16 + p] = v;
    }
}

// ---------------------------------------------------------------------------
// agree-GEMM: G[(i,s)][(o,p)] = sum_b xt[(i,s)][b] * vhT[(o,p)][b]  (K=512),
// epilogue contracts with W in-register and updates bij (unique writer).
// grid = 576 blocks (i-pairs), block = 320 (wave w -> o = w).
// ---------------------------------------------------------------------------
__global__ void k_agree(const _Float16* __restrict__ xt,
                        const _Float16* __restrict__ vhT,
                        const float* __restrict__ W,
                        float* __restrict__ bij) {
    const int ipair = blockIdx.x;          // rows (i0..i0+1), i0 = 2*ipair
    const int wave  = threadIdx.x >> 5;    // o
    const int lane  = threadIdx.x & 31;
    const int nm    = lane & 15;
    const int khalf = (lane >> 4) * 8;

    const _Float16* arow = xt  + (size_t)(ipair * 16 + nm) * BATCH + khalf;
    const _Float16* brow = vhT + (size_t)(wave  * 16 + nm) * BATCH + khalf;

    v8f acc = {};
    for (int ks = 0; ks < BATCH / 32; ++ks) {
        v16h a = load_frag(arow);
        v16h b = load_frag(brow);
        acc = __builtin_amdgcn_wmma_f32_16x16x32_f16(
            false, a, false, b, (short)0, acc, false, false);
        arow += 32; brow += 32;
    }

    // VGPR r, lanes 0-15: G row = r (=> i_local 0, s=r); lanes 16-31: row 8+r
    // (=> i_local 1, s=r). Column = (o=wave, p = lane&15).
    const int ilocal = lane >> 4;
    const int p = lane & 15;
    const int i = ipair * 2 + ilocal;
    float sum = 0.0f;
#pragma unroll
    for (int r = 0; r < 8; ++r) {
        float w = W[(((size_t)i * OUT_CAPS + wave) * P_DIM + p) * S_DIM + r];
        sum += w * acc[r];
    }
    // reduce over p (16 lanes within each half-wave)
    for (int off = 1; off < 16; off <<= 1)
        sum += __shfl_xor(sum, off, 32);
    if (p == 0)
        bij[(size_t)i * OUT_CAPS + wave] += sum * (1.0f / (float)BATCH);
}

// ---------------------------------------------------------------------------
extern "C" void kernel_launch(void* const* d_in, const int* in_sizes, int n_in,
                              void* d_out, int out_size, void* d_ws, size_t ws_size,
                              hipStream_t stream) {
    const float* x = (const float*)d_in[0];   // (512,1152,8)
    const float* W = (const float*)d_in[1];   // (1152,10,16,8)
    float* vout    = (float*)d_out;           // (512,10,16)

    // workspace carve-up (256B aligned)
    char* ws = (char*)d_ws;
    size_t off = 0;
    auto carve = [&](size_t bytes) { char* p = ws + off; off = (off + bytes + 255) & ~(size_t)255; return p; };
    float*    bij   = (float*)   carve((size_t)IN_CAPS * OUT_CAPS * 4);       // 46 KB
    float*    c     = (float*)   carve((size_t)IN_CAPS * OUT_CAPS * 4);       // 46 KB
    _Float16* cwT   = (_Float16*)carve((size_t)NTOT * KTOT * 2);              // 2.9 MB
    _Float16* xh    = (_Float16*)carve((size_t)BATCH * KTOT * 2);             // 9.4 MB
    _Float16* xt    = (_Float16*)carve((size_t)KTOT * BATCH * 2);             // 9.4 MB
    float*    spart = (float*)   carve((size_t)KSPLIT * BATCH * NTOT * 4);    // 2.6 MB
    float*    s     = (float*)   carve((size_t)BATCH * NTOT * 4);             // 0.3 MB
    _Float16* vhT   = (_Float16*)carve((size_t)NTOT * BATCH * 2);             // 0.2 MB
    (void)ws_size; (void)in_sizes; (void)n_in; (void)out_size;

    // one-time per call: f16 conversions + zero routing logits
    {
        size_t n = (size_t)BATCH * KTOT;
        k_convert_x<<<dim3((unsigned)((n + 255) / 256)), 256, 0, stream>>>(x, xh, xt);
        k_zero<<<dim3((IN_CAPS * OUT_CAPS + 255) / 256), 256, 0, stream>>>(bij, IN_CAPS * OUT_CAPS);
    }

    for (int it = 0; it < 3; ++it) {
        // c = softmax_i(bij)  (it==0: bij==0 -> uniform, matches reference)
        k_softmax<<<dim3(OUT_CAPS), 256, 0, stream>>>(bij, c);
        // cwT = f16(c * W)
        {
            size_t n = (size_t)NTOT * KTOT;
            k_prep_cw<<<dim3((unsigned)((n + 255) / 256)), 256, 0, stream>>>(c, W, cwT);
        }
        // s = x @ cW   (split-K, deterministic)
        k_s_gemm<<<dim3(BATCH / 16, KSPLIT), 320, 0, stream>>>(xh, cwT, spart);
        k_reduce_s<<<dim3((BATCH * NTOT + 255) / 256), 256, 0, stream>>>(spart, s);
        // v = squash(s)
        k_squash<<<dim3((BATCH * OUT_CAPS + 255) / 256), 256, 0, stream>>>(
            s, vhT, vout, (it == 2) ? 1 : 0);
        // bij += mean_b <u_hat, v>   (skip after last iteration)
        if (it < 2)
            k_agree<<<dim3(IN_CAPS / 2), 320, 0, stream>>>(xt, vhT, W, bij);
    }
}
